// RoleAllocation_7773890806138
// MI455X (gfx1250) — matrix operand order repaired and checked by reading, not compiled, not run on hardware
//
#include <hip/hip_runtime.h>
#include <hip/hip_bf16.h>
#include <math.h>

typedef __bf16 v16bf __attribute__((ext_vector_type(16)));
typedef __bf16 v4bf  __attribute__((ext_vector_type(4)));
typedef float  v8f   __attribute__((ext_vector_type(8)));

#define NQ_   256
#define NR_   1024
#define DD_   384
#define HH_   64
#define CC_   128
#define KK_   8

#define LOG_VAR2_ (-4.605170185988091f)  /* ln(0.01) */
#define LN_EPS_   (1e-5f)

/* workspace byte offsets */
#define WS_W1   0          /* 12x4 tiles * 512 bf16 = 49152 B */
#define WS_W21  49152      /* 2x4 tiles  * 512 bf16 = 8192 B  */
#define WS_W22  57344
#define WS_W3   65536
#define WS_W4   73728      /* 2x24 tiles * 512 bf16 = 49152 B */
#define WS_REMB 131072     /* 262144*64 bf16 = 33554432 B */

/* output float offsets */
#define OUT_IDX  0
#define OUT_LP   2048
#define OUT_SUM  2304
#define OUT_LOSS 18688
#define OUT_N    18689

static __device__ __forceinline__ float bf_lo(unsigned u) {
  return __uint_as_float(u << 16);
}
static __device__ __forceinline__ float bf_hi(unsigned u) {
  return __uint_as_float(u & 0xffff0000u);
}

static __device__ __forceinline__ void pack_B(const float* W, __bf16* dst,
                                              int Kin, int Nout, int tid, int nt) {
  const int TN = Nout / 16, TK = Kin / 32;
  const int total = TK * TN * 512;
  for (int p = tid; p < total; p += nt) {
    int e = p & 15;
    int L = (p >> 4) & 31;
    int t = p >> 9;
    int t_n = t % TN, t_k = t / TN;
    int n = t_n * 16 + (L & 15);
    int k = t_k * 32 + ((L >= 16) ? 8 : 0) + ((e < 8) ? e : 16 + (e - 8));
    dst[p] = (__bf16)W[n * Kin + k];
  }
}

__global__ void prep_kernel(const float* w1, const float* w21, const float* w22,
                            const float* w3, const float* w4, unsigned char* ws) {
  int tid = blockIdx.x * blockDim.x + threadIdx.x;
  int nt  = gridDim.x * blockDim.x;
  pack_B(w1,  (__bf16*)(ws + WS_W1),  384, 64,  tid, nt);
  pack_B(w21, (__bf16*)(ws + WS_W21), 64,  64,  tid, nt);
  pack_B(w22, (__bf16*)(ws + WS_W22), 64,  64,  tid, nt);
  pack_B(w3,  (__bf16*)(ws + WS_W3),  64,  64,  tid, nt);
  pack_B(w4,  (__bf16*)(ws + WS_W4),  64,  384, tid, nt);
}

__global__ void zero_kernel(float* out, int n) {
  int i = blockIdx.x * blockDim.x + threadIdx.x;
  if (i < n) out[i] = 0.0f;
}

#define WMMA_BF16(A, B, C) \
  __builtin_amdgcn_wmma_f32_16x16x32_bf16(false, (A), false, (B), (short)0, (C), false, false)

__global__ __launch_bounds__(256)
void vae_kernel(const float* __restrict__ roles, const float* __restrict__ eps,
                const float* __restrict__ b1, const float* __restrict__ b21,
                const float* __restrict__ b22, const float* __restrict__ b3,
                const float* __restrict__ b4,
                const unsigned char* __restrict__ ws_ro,
                __bf16* __restrict__ role_emb, float* __restrict__ loss_out) {
  __shared__ __align__(16) float stage[8][16 * 64];  /* per-wave 16x64 f32, 32 KB */
  __shared__ float rowscale[8][16];

  const int w    = threadIdx.x >> 5;
  const int lane = threadIdx.x & 31;
  const int half = (lane >= 16) ? 1 : 0;
  const int ln   = lane & 15;
  const size_t row_base = (size_t)blockIdx.x * 128 + (size_t)w * 16;
  float* st = &stage[w][0];

  /* ---------- layer 1: h = relu(roles @ W1^T + b1), K=384 ---------- */
  v8f acc[4];
#pragma unroll
  for (int t = 0; t < 4; ++t) acc[t] = (v8f){};
  const float*  arow = roles + (row_base + (size_t)ln) * DD_;
  const __bf16* w1p  = (const __bf16*)(ws_ro + WS_W1);
#pragma unroll
  for (int kst = 0; kst < 12; ++kst) {
    v16bf a;
    const int kb = kst * 32 + half * 8;
#pragma unroll
    for (int v = 0; v < 8; ++v) {
      int kk = kb + ((v < 4) ? 2 * v : 16 + 2 * (v - 4));
      a[2 * v]     = (__bf16)arow[kk];
      a[2 * v + 1] = (__bf16)arow[kk + 1];
    }
#pragma unroll
    for (int tn = 0; tn < 4; ++tn) {
      v16bf b = *(const v16bf*)(w1p + ((size_t)(kst * 4 + tn) * 32 + lane) * 16);
      acc[tn] = WMMA_BF16(a, b, acc[tn]);
    }
  }
#pragma unroll
  for (int tn = 0; tn < 4; ++tn) {
    int n = ln + 16 * tn;
    float bb = b1[n];
#pragma unroll
    for (int i = 0; i < 8; ++i) {
      float hv = acc[tn][i] + bb;
      st[(i + 8 * half) * 64 + n] = fmaxf(hv, 0.0f);
    }
  }
  __syncthreads();

  /* A fragments of h (K=64 -> two 32-wide k-steps) */
  v16bf ah0, ah1;
#pragma unroll
  for (int v = 0; v < 8; ++v) {
    int kk = half * 8 + ((v < 4) ? 2 * v : 16 + 2 * (v - 4));
    ah0[2 * v]     = (__bf16)st[ln * 64 + kk];
    ah0[2 * v + 1] = (__bf16)st[ln * 64 + kk + 1];
    ah1[2 * v]     = (__bf16)st[ln * 64 + 32 + kk];
    ah1[2 * v + 1] = (__bf16)st[ln * 64 + 32 + kk + 1];
  }

  /* ---------- mu / log_var ---------- */
  const __bf16* w21p = (const __bf16*)(ws_ro + WS_W21);
  const __bf16* w22p = (const __bf16*)(ws_ro + WS_W22);
  v8f amu[4], alv[4];
#pragma unroll
  for (int t = 0; t < 4; ++t) { amu[t] = (v8f){}; alv[t] = (v8f){}; }
#pragma unroll
  for (int tn = 0; tn < 4; ++tn) {
    v16bf bm0 = *(const v16bf*)(w21p + ((size_t)(0 * 4 + tn) * 32 + lane) * 16);
    v16bf bm1 = *(const v16bf*)(w21p + ((size_t)(1 * 4 + tn) * 32 + lane) * 16);
    amu[tn] = WMMA_BF16(ah0, bm0, amu[tn]);
    amu[tn] = WMMA_BF16(ah1, bm1, amu[tn]);
    v16bf bl0 = *(const v16bf*)(w22p + ((size_t)(0 * 4 + tn) * 32 + lane) * 16);
    v16bf bl1 = *(const v16bf*)(w22p + ((size_t)(1 * 4 + tn) * 32 + lane) * 16);
    alv[tn] = WMMA_BF16(ah0, bl0, alv[tn]);
    alv[tn] = WMMA_BF16(ah1, bl1, alv[tn]);
  }

  /* ---------- z, kld, row sumsq ---------- */
  float zreg[4][8];
  float kld_part = 0.0f;
  float rowss[8];
#pragma unroll
  for (int i = 0; i < 8; ++i) rowss[i] = 0.0f;
#pragma unroll
  for (int tn = 0; tn < 4; ++tn) {
    int n = ln + 16 * tn;
    float bmu = b21[n], blv = b22[n];
#pragma unroll
    for (int i = 0; i < 8; ++i) {
      size_t row = row_base + i + 8 * half;
      float mu = amu[tn][i] + bmu;
      float lv = alv[tn][i] + blv;
      float ev = eps[row * HH_ + n];
      float zz = mu + ev * __expf(0.5f * lv) * 0.1f;
      zreg[tn][i] = zz;
      kld_part += 1.0f - LOG_VAR2_ + lv - (mu * mu + __expf(lv)) * 100.0f;
      rowss[i] += zz * zz;
    }
  }
  /* reduce sumsq across the 16 lanes holding each row -> 1/max(||z||,eps) */
#pragma unroll
  for (int i = 0; i < 8; ++i) {
    float s = rowss[i];
    s += __shfl_xor(s, 1, 32);
    s += __shfl_xor(s, 2, 32);
    s += __shfl_xor(s, 4, 32);
    s += __shfl_xor(s, 8, 32);
    rowss[i] = 1.0f / fmaxf(sqrtf(s), 1e-12f);
  }
  /* stage raw z (for next GEMM) + per-row scales */
  __syncthreads();
#pragma unroll
  for (int tn = 0; tn < 4; ++tn) {
    int n = ln + 16 * tn;
#pragma unroll
    for (int i = 0; i < 8; ++i)
      st[(i + 8 * half) * 64 + n] = zreg[tn][i];
  }
  if (ln == 0) {
#pragma unroll
    for (int i = 0; i < 8; ++i) rowscale[w][i + 8 * half] = rowss[i];
  }
  __syncthreads();

  /* coalesced role_emb store: per c, wave writes 256 contiguous bytes */
#pragma unroll
  for (int c = 0; c < 8; ++c) {
    int m   = 2 * c + half;       /* row within tile */
    int col = ln * 4;             /* 4 consecutive columns */
    float s = rowscale[w][m];
    const float4 f = *(const float4*)&st[m * 64 + col];
    v4bf o;
    o[0] = (__bf16)(f.x * s);
    o[1] = (__bf16)(f.y * s);
    o[2] = (__bf16)(f.z * s);
    o[3] = (__bf16)(f.w * s);
    *(v4bf*)(role_emb + (row_base + m) * HH_ + col) = o;
  }

  /* ---------- h2 = relu(z @ W3^T + b3) ---------- */
  v16bf az0, az1;
#pragma unroll
  for (int v = 0; v < 8; ++v) {
    int kk = half * 8 + ((v < 4) ? 2 * v : 16 + 2 * (v - 4));
    az0[2 * v]     = (__bf16)st[ln * 64 + kk];
    az0[2 * v + 1] = (__bf16)st[ln * 64 + kk + 1];
    az1[2 * v]     = (__bf16)st[ln * 64 + 32 + kk];
    az1[2 * v + 1] = (__bf16)st[ln * 64 + 32 + kk + 1];
  }
  const __bf16* w3p = (const __bf16*)(ws_ro + WS_W3);
  v8f ah2[4];
#pragma unroll
  for (int t = 0; t < 4; ++t) ah2[t] = (v8f){};
#pragma unroll
  for (int tn = 0; tn < 4; ++tn) {
    v16bf b0  = *(const v16bf*)(w3p + ((size_t)(0 * 4 + tn) * 32 + lane) * 16);
    v16bf b1v = *(const v16bf*)(w3p + ((size_t)(1 * 4 + tn) * 32 + lane) * 16);
    ah2[tn] = WMMA_BF16(az0, b0, ah2[tn]);
    ah2[tn] = WMMA_BF16(az1, b1v, ah2[tn]);
  }
  __syncthreads();
#pragma unroll
  for (int tn = 0; tn < 4; ++tn) {
    int n = ln + 16 * tn;
    float bb = b3[n];
#pragma unroll
    for (int i = 0; i < 8; ++i)
      st[(i + 8 * half) * 64 + n] = fmaxf(ah2[tn][i] + bb, 0.0f);
  }
  __syncthreads();

  v16bf ax0, ax1;
#pragma unroll
  for (int v = 0; v < 8; ++v) {
    int kk = half * 8 + ((v < 4) ? 2 * v : 16 + 2 * (v - 4));
    ax0[2 * v]     = (__bf16)st[ln * 64 + kk];
    ax0[2 * v + 1] = (__bf16)st[ln * 64 + kk + 1];
    ax1[2 * v]     = (__bf16)st[ln * 64 + 32 + kk];
    ax1[2 * v + 1] = (__bf16)st[ln * 64 + 32 + kk + 1];
  }

  /* ---------- x_hat = h2 @ W4^T + b4 ; mse ---------- */
  const __bf16* w4p = (const __bf16*)(ws_ro + WS_W4);
  float mse_part = 0.0f;
#pragma unroll 4
  for (int tn = 0; tn < 24; ++tn) {
    v8f accx = (v8f){};
    v16bf b0  = *(const v16bf*)(w4p + ((size_t)(0 * 24 + tn) * 32 + lane) * 16);
    v16bf b1v = *(const v16bf*)(w4p + ((size_t)(1 * 24 + tn) * 32 + lane) * 16);
    accx = WMMA_BF16(ax0, b0, accx);
    accx = WMMA_BF16(ax1, b1v, accx);
    int d = ln + 16 * tn;
    float bb = b4[d];
#pragma unroll
    for (int i = 0; i < 8; ++i) {
      size_t row = row_base + i + 8 * half;
      float diff = (accx[i] + bb) - roles[row * DD_ + d];
      mse_part += diff * diff;
    }
  }

  /* ---------- loss reduction ---------- */
  float total = mse_part * (1.0f / ((float)NQ_ * NR_ * DD_))
              + kld_part * (-0.5f / ((float)NQ_ * NR_ * HH_));
  total += __shfl_xor(total, 1, 32);
  total += __shfl_xor(total, 2, 32);
  total += __shfl_xor(total, 4, 32);
  total += __shfl_xor(total, 8, 32);
  total += __shfl_xor(total, 16, 32);
  if (lane == 0) atomicAdd(loss_out, total);
}

/* one block per q; sequential K=8 scan inside */
__global__ __launch_bounds__(256)
void scan_kernel(const float* __restrict__ contexts, const float* __restrict__ rand_u,
                 const float* __restrict__ wc, const float* __restrict__ bc,
                 const float* __restrict__ init_emb, const __bf16* __restrict__ role_emb,
                 float* __restrict__ out_idx, float* __restrict__ out_lp,
                 float* __restrict__ out_sum) {
  const int q = blockIdx.x;
  const int tid = threadIdx.x;
  __shared__ __align__(16) float ctx[128];
  __shared__ float hist[64], cur[64], ce[64];
  __shared__ float sred[256];
  __shared__ float sscan[256];
  __shared__ int s_idx;
  __shared__ float s_mean, s_var, s_scale, s_total, s_u, s_lp;

  /* async copy contexts row (512B) into LDS: one wave-wide b128 async load */
  if (tid < 32) {
    unsigned ldsa = (unsigned)(uintptr_t)&ctx[0] + (unsigned)tid * 16u;
    unsigned voff = (unsigned)tid * 16u;
    unsigned long long sbase = (unsigned long long)(uintptr_t)(contexts + (size_t)q * CC_);
    asm volatile("global_load_async_to_lds_b128 %0, %1, %2"
                 :: "v"(ldsa), "v"(voff), "s"(sbase) : "memory");
    asm volatile("s_wait_asynccnt 0" ::: "memory");
  }
  if (tid < 64) { hist[tid] = init_emb[tid]; cur[tid] = init_emb[tid]; }
  if (tid == 0) s_lp = 0.0f;
  __syncthreads();

  const __bf16* remb = role_emb + (size_t)q * NR_ * HH_;

  for (int k = 0; k < KK_; ++k) {
    /* hist = layer_norm(hist + cur) */
    if (tid < 64) sred[tid] = hist[tid] + cur[tid];
    __syncthreads();
    if (tid == 0) {
      float m = 0.0f;
      for (int j = 0; j < 64; ++j) m += sred[j];
      m *= (1.0f / 64.0f);
      float v = 0.0f;
      for (int j = 0; j < 64; ++j) { float d = sred[j] - m; v += d * d; }
      s_mean = m; s_var = v * (1.0f / 64.0f);
    }
    __syncthreads();
    if (tid < 64) hist[tid] = (sred[tid] - s_mean) * rsqrtf(s_var + LN_EPS_);
    __syncthreads();

    /* ce = l2norm(concat(ctx, hist) @ wc^T + bc) */
    if (tid < 64) {
      float s = bc[tid];
      const float* wr = wc + tid * (CC_ + HH_);
      for (int c = 0; c < CC_; ++c) s += ctx[c] * wr[c];
      for (int j = 0; j < 64; ++j) s += hist[j] * wr[CC_ + j];
      ce[tid] = s;
    }
    __syncthreads();
    if (tid == 0) {
      float ss = 0.0f;
      for (int j = 0; j < 64; ++j) ss += ce[j] * ce[j];
      s_scale = 1.0f / fmaxf(sqrtf(ss), 1e-12f);
      s_u = rand_u[q * KK_ + k];
      s_idx = NR_ - 1;
    }
    __syncthreads();
    if (tid < 64) ce[tid] *= s_scale;
    __syncthreads();

    /* scores: 4 consecutive r per thread; 128B vector loads of bf16 rows */
    const int r0 = tid * 4;
    float sc[4];
#pragma unroll
    for (int j = 0; j < 4; ++j) {
      const uint4* rp4 = (const uint4*)(remb + (size_t)(r0 + j) * HH_);
      float s = 0.0f;
#pragma unroll
      for (int t4 = 0; t4 < 8; ++t4) {
        uint4 u = rp4[t4];
        int hb = t4 * 8;
        s += ce[hb + 0] * bf_lo(u.x) + ce[hb + 1] * bf_hi(u.x);
        s += ce[hb + 2] * bf_lo(u.y) + ce[hb + 3] * bf_hi(u.y);
        s += ce[hb + 4] * bf_lo(u.z) + ce[hb + 5] * bf_hi(u.z);
        s += ce[hb + 6] * bf_lo(u.w) + ce[hb + 7] * bf_hi(u.w);
      }
      sc[j] = s;
    }
    float lm = fmaxf(fmaxf(sc[0], sc[1]), fmaxf(sc[2], sc[3]));
    sred[tid] = lm;
    __syncthreads();
    for (int stp = 128; stp > 0; stp >>= 1) {
      if (tid < stp) sred[tid] = fmaxf(sred[tid], sred[tid + stp]);
      __syncthreads();
    }
    float mx = sred[0];
    __syncthreads();

    float p[4];
    float ts = 0.0f;
#pragma unroll
    for (int j = 0; j < 4; ++j) { p[j] = __expf(sc[j] - mx); ts += p[j]; }
    sscan[tid] = ts;
    __syncthreads();
    if (tid == 0) {
      float run = 0.0f;
      for (int t = 0; t < 256; ++t) { float v = sscan[t]; sscan[t] = run; run += v; }
      s_total = run;
    }
    __syncthreads();

    float thr = s_u * s_total;
    float run = sscan[tid];
    int cand = 0x7fffffff;
#pragma unroll
    for (int j = 0; j < 4; ++j) {
      run += p[j];
      if (run > thr && cand == 0x7fffffff) cand = r0 + j;
    }
    if (cand != 0x7fffffff) atomicMin(&s_idx, cand);
    __syncthreads();
    const int idx = s_idx;

    if (tid == 0) {
      const uint4* rp4 = (const uint4*)(remb + (size_t)idx * HH_);
      float s = 0.0f;
      for (int t4 = 0; t4 < 8; ++t4) {
        uint4 u = rp4[t4];
        int hb = t4 * 8;
        s += ce[hb + 0] * bf_lo(u.x) + ce[hb + 1] * bf_hi(u.x);
        s += ce[hb + 2] * bf_lo(u.y) + ce[hb + 3] * bf_hi(u.y);
        s += ce[hb + 4] * bf_lo(u.z) + ce[hb + 5] * bf_hi(u.z);
        s += ce[hb + 6] * bf_lo(u.w) + ce[hb + 7] * bf_hi(u.w);
      }
      s_lp += (s - mx) - __logf(s_total);
      out_idx[q * KK_ + k] = (float)idx;
    }
    if (tid < 64) cur[tid] = (float)remb[(size_t)idx * HH_ + tid];
    __syncthreads();
  }
  if (tid < 64) out_sum[q * HH_ + tid] = hist[tid];
  if (tid == 0) out_lp[q] = s_lp;
}

extern "C" void kernel_launch(void* const* d_in, const int* in_sizes, int n_in,
                              void* d_out, int out_size, void* d_ws, size_t ws_size,
                              hipStream_t stream) {
  const float* roles    = (const float*)d_in[0];
  const float* contexts = (const float*)d_in[1];
  const float* eps      = (const float*)d_in[2];
  const float* rand_u   = (const float*)d_in[3];
  const float* w1  = (const float*)d_in[4];
  const float* b1  = (const float*)d_in[5];
  const float* w21 = (const float*)d_in[6];
  const float* b21 = (const float*)d_in[7];
  const float* w22 = (const float*)d_in[8];
  const float* b22 = (const float*)d_in[9];
  const float* w3  = (const float*)d_in[10];
  const float* b3  = (const float*)d_in[11];
  const float* w4  = (const float*)d_in[12];
  const float* b4  = (const float*)d_in[13];
  const float* wc  = (const float*)d_in[14];
  const float* bc  = (const float*)d_in[15];
  const float* init_emb = (const float*)d_in[16];

  float* out = (float*)d_out;
  unsigned char* ws = (unsigned char*)d_ws;
  __bf16* role_emb = (__bf16*)(ws + WS_REMB);

  zero_kernel<<<(OUT_N + 255) / 256, 256, 0, stream>>>(out, OUT_N);
  prep_kernel<<<64, 256, 0, stream>>>(w1, w21, w22, w3, w4, ws);
  vae_kernel<<<(NQ_ * NR_) / 128, 256, 0, stream>>>(
      roles, eps, b1, b21, b22, b3, b4, ws, role_emb, out + OUT_LOSS);
  scan_kernel<<<NQ_, 256, 0, stream>>>(
      contexts, rand_u, wc, bc, init_emb, role_emb,
      out + OUT_IDX, out + OUT_LP, out + OUT_SUM);
}
